// RegionAttention_14783277432912
// MI455X (gfx1250) — matrix operand order repaired and checked by reading, not compiled, not run on hardware
//
#include <hip/hip_runtime.h>
#include <stdint.h>

typedef _Float16 v16h __attribute__((ext_vector_type(16)));
typedef _Float16 v8h  __attribute__((ext_vector_type(8)));
typedef float    v8f  __attribute__((ext_vector_type(8)));
typedef float    v4f  __attribute__((ext_vector_type(4)));

#define WB    4
#define WR    2048
#define WD    1024
#define WH    16
#define WDH   64
#define WDOUT 1024

// ---------------------------------------------------------------------------
// CDNA5 WMMA fragment loaders (wave32).
// A 16x32 f16: lanes 0-15 row=lane, K chunks {0..7,16..23}; lanes 16-31 same
// rows, K chunks {8..15,24..31}.  -> two 16B LDS loads per lane.
// B 32x16 f16: lane column n=lane&15; K chunks same as A ({0..7,16..23} / {8..15,24..31}).
// ---------------------------------------------------------------------------
__device__ __forceinline__ v16h frag_a(const _Float16* rowp, int lh) {
  union { v16h v; v8h h[2]; } u;
  u.h[0] = *(const v8h*)(rowp + lh * 8);
  u.h[1] = *(const v8h*)(rowp + 16 + lh * 8);
  return u.v;
}
__device__ __forceinline__ v16h frag_b(const _Float16* colp, int lh) {
  union { v16h v; v8h h[2]; } u;
  u.h[0] = *(const v8h*)(colp + lh * 8);
  u.h[1] = *(const v8h*)(colp + 16 + lh * 8);
  return u.v;
}
__device__ __forceinline__ v8f wmma_f16(v16h a, v16h b, v8f c) {
  return __builtin_amdgcn_wmma_f32_16x16x32_f16(false, a, false, b,
                                                (short)0, c, false, false);
}
// CDNA5 async global->LDS copy (ASYNCcnt-tracked); dst = LDS byte offset VGPR.
__device__ __forceinline__ void async_ld_b128(const _Float16* lds_dst,
                                              const _Float16* gsrc) {
  uint32_t loff = (uint32_t)(uintptr_t)lds_dst;     // low 32 bits = LDS addr
  uint64_t gaddr = (uint64_t)(uintptr_t)gsrc;
  asm volatile("global_load_async_to_lds_b128 %0, %1, off"
               :: "v"(loff), "v"(gaddr) : "memory");
}
__device__ __forceinline__ void wait_async0() {
  asm volatile("s_wait_asynccnt 0x0" ::: "memory");
}
__device__ __forceinline__ float redmax16(float v) {
  v = fmaxf(v, __shfl_xor(v, 1));
  v = fmaxf(v, __shfl_xor(v, 2));
  v = fmaxf(v, __shfl_xor(v, 4));
  v = fmaxf(v, __shfl_xor(v, 8));
  return v;
}
__device__ __forceinline__ float redsum16(float v) {
  v += __shfl_xor(v, 1);
  v += __shfl_xor(v, 2);
  v += __shfl_xor(v, 4);
  v += __shfl_xor(v, 8);
  return v;
}

// ---------------------------------------------------------------------------
// Generic C[M,N] = A[M,K] @ W[K,N] + bias.  A: f32 or f16; W/bias f32.
// Workgroup tile 128x128, K-step 64. 8 waves, each wave 32x64 (2x4 wmma tiles)
// f16-A path stages A via async global->LDS copies.
// ---------------------------------------------------------------------------
template <typename AT, bool OUT_HALF>
__global__ __launch_bounds__(256)
void gemm_wmma(const AT* __restrict__ A, const float* __restrict__ W,
               const float* __restrict__ bias, void* __restrict__ Cout,
               int M, int N, int K) {
  constexpr int AS = 72;                       // padded LDS row stride (halfs)
  __shared__ _Float16 As[128 * AS];            // A tile   [m][k]  (k = 64)
  __shared__ _Float16 Bs[128 * AS];            // W^T tile [n][k]
  const int t    = threadIdx.x;
  const int lane = t & 31;
  const int wave = t >> 5;
  const int lh   = lane >> 4;
  const int col  = lane & 15;
  const int rh   = lh * 8;
  const int m0   = blockIdx.y * 128;
  const int n0   = blockIdx.x * 128;
  const int wm   = (wave & 3) * 32;
  const int wn   = (wave >> 2) * 64;

  v8f zero8;
#pragma unroll
  for (int e = 0; e < 8; ++e) zero8[e] = 0.0f;
  v8f acc[2][4];
#pragma unroll
  for (int i = 0; i < 2; ++i)
#pragma unroll
    for (int j = 0; j < 4; ++j) acc[i][j] = zero8;

  for (int kt = 0; kt < K; kt += 64) {
    __syncthreads();
    // ---- stage A tile ----
    if constexpr (sizeof(AT) == 4) {
#pragma unroll
      for (int i = 0; i < 8; ++i) {
        int f = t + i * 256;                       // 2048 float4 = 128r x 16c
        int row = f >> 4, kq = (f & 15) * 4;
        v4f a = *(const v4f*)((const float*)A + (size_t)(m0 + row) * K + kt + kq);
        _Float16* dst = &As[row * AS + kq];
        dst[0] = (_Float16)a.x; dst[1] = (_Float16)a.y;
        dst[2] = (_Float16)a.z; dst[3] = (_Float16)a.w;
      }
    } else {
#pragma unroll
      for (int i = 0; i < 4; ++i) {
        int f = t + i * 256;                       // 1024 v8h = 128r x 8c
        int row = f >> 3, kq = (f & 7) * 8;
        async_ld_b128(&As[row * AS + kq],
                      (const _Float16*)A + (size_t)(m0 + row) * K + kt + kq);
      }
    }
    // ---- stage W^T tile ----
#pragma unroll
    for (int i = 0; i < 8; ++i) {
      int f = t + i * 256;                         // 2048 float4 = 64k x 32c
      int kr = f >> 5, nq = (f & 31) * 4;
      v4f w = *(const v4f*)(W + (size_t)(kt + kr) * N + n0 + nq);
      Bs[(nq + 0) * AS + kr] = (_Float16)w.x;
      Bs[(nq + 1) * AS + kr] = (_Float16)w.y;
      Bs[(nq + 2) * AS + kr] = (_Float16)w.z;
      Bs[(nq + 3) * AS + kr] = (_Float16)w.w;
    }
    // speculative prefetch of next W tile (L2 hint)
    if (kt + 64 < K) {
      __builtin_prefetch(W + (size_t)(kt + 64 + (t >> 2)) * N + n0 + (t & 3) * 32, 0, 1);
    }
    if constexpr (sizeof(AT) == 2) wait_async0();
    __syncthreads();

    v16h af[2][2], bf[4][2];
#pragma unroll
    for (int i = 0; i < 2; ++i)
#pragma unroll
      for (int kk = 0; kk < 2; ++kk)
        af[i][kk] = frag_a(&As[(wm + i * 16 + col) * AS + kk * 32], lh);
#pragma unroll
    for (int j = 0; j < 4; ++j)
#pragma unroll
      for (int kk = 0; kk < 2; ++kk)
        bf[j][kk] = frag_b(&Bs[(wn + j * 16 + col) * AS + kk * 32], lh);
#pragma unroll
    for (int kk = 0; kk < 2; ++kk)
#pragma unroll
      for (int i = 0; i < 2; ++i)
#pragma unroll
        for (int j = 0; j < 4; ++j)
          acc[i][j] = wmma_f16(af[i][kk], bf[j][kk], acc[i][j]);
  }

  // ---- epilogue: bias + store (C/D layout: row = i + rh, col = lane&15) ----
#pragma unroll
  for (int i = 0; i < 2; ++i) {
#pragma unroll
    for (int j = 0; j < 4; ++j) {
      int n = n0 + wn + j * 16 + col;
      float bvv = bias[n];
#pragma unroll
      for (int e = 0; e < 8; ++e) {
        int m = m0 + wm + i * 16 + e + rh;
        float v = acc[i][j][e] + bvv;
        if constexpr (OUT_HALF)
          ((_Float16*)Cout)[(size_t)m * N + n] = (_Float16)v;
        else
          ((float*)Cout)[(size_t)m * N + n] = v;
      }
    }
  }
}

// ---------------------------------------------------------------------------
// Flash attention: per (b,h) and 128-row q tile; stream 64-wide K/V blocks.
// scores = (Q K^T) * sqrt(D); mask==0 -> -big; online softmax; O += P V.
// Q and K tiles staged via async global->LDS; V needs a transpose so it goes
// through VGPRs.  qh/kh/vh/oh layout: [b][r][h*DH + dh]  (f16)
// ---------------------------------------------------------------------------
__global__ __launch_bounds__(256)
void attn_kernel(const _Float16* __restrict__ qh, const _Float16* __restrict__ kh,
                 const _Float16* __restrict__ vh, const int* __restrict__ mask,
                 _Float16* __restrict__ oh) {
  constexpr int QS = 72, KS = 72, VS = 72, PS = 72;
  __shared__ _Float16 Qs[128 * QS];     // [q][dh]
  __shared__ _Float16 Ks[64 * KS];      // [s][dh]   (B-frag: lane = s col)
  __shared__ _Float16 Vts[64 * VS];     // [dh][s]   (B-frag: lane = dh col)
  __shared__ _Float16 Ps[128 * PS];     // per-wave 16 rows [q][s]
  const int t    = threadIdx.x;
  const int lane = t & 31;
  const int wave = t >> 5;
  const int lh   = lane >> 4;
  const int col  = lane & 15;
  const int rh   = lh * 8;
  const int bh   = blockIdx.y;
  const int b    = bh >> 4;             // H = 16
  const int h    = bh & 15;
  const int q0   = blockIdx.x * 128;
  const size_t base = ((size_t)b * WR) * WD + (size_t)h * WDH;

  // ---- load Q tile asynchronously (first loop iteration waits for it) ----
#pragma unroll
  for (int i = 0; i < 4; ++i) {
    int f = t + i * 256;                          // 1024 v8h = 128r x 8c
    int qr = f >> 3, dq = (f & 7) * 8;
    async_ld_b128(&Qs[qr * QS + dq], qh + base + (size_t)(q0 + qr) * WD + dq);
  }

  v8f zero8;
#pragma unroll
  for (int e = 0; e < 8; ++e) zero8[e] = 0.0f;
  float m_i[8], l_i[8];
  v8f o[4];
#pragma unroll
  for (int e = 0; e < 8; ++e) { m_i[e] = -3.0e38f; l_i[e] = 0.0f; }
#pragma unroll
  for (int j = 0; j < 4; ++j) o[j] = zero8;

  const int qw = wave * 16;

  for (int s0 = 0; s0 < WR; s0 += 64) {
    __syncthreads();
    // ---- stage K block row-major via async copy ----
#pragma unroll
    for (int i = 0; i < 2; ++i) {
      int f = t + i * 256;                        // 512 v8h = 64r x 8c
      int s = f >> 3, dq = (f & 7) * 8;
      async_ld_b128(&Ks[s * KS + dq], kh + base + (size_t)(s0 + s) * WD + dq);
    }
    // ---- stage V block transposed (VGPR-mediated) ----
#pragma unroll
    for (int i = 0; i < 2; ++i) {
      int f = t + i * 256;
      int s = f >> 3, dq = (f & 7) * 8;
      v8h v = *(const v8h*)(vh + base + (size_t)(s0 + s) * WD + dq);
#pragma unroll
      for (int e = 0; e < 8; ++e) Vts[(dq + e) * VS + s] = v[e];
    }
    wait_async0();
    __syncthreads();

    // ---- S = Q K^T ----
    v16h qa0 = frag_a(&Qs[(qw + col) * QS + 0], lh);
    v16h qa1 = frag_a(&Qs[(qw + col) * QS + 32], lh);
    v8f S[4];
#pragma unroll
    for (int nt = 0; nt < 4; ++nt) {
      v16h kb0 = frag_b(&Ks[(nt * 16 + col) * KS + 0], lh);
      v16h kb1 = frag_b(&Ks[(nt * 16 + col) * KS + 32], lh);
      v8f s_acc = zero8;
      s_acc = wmma_f16(qa0, kb0, s_acc);
      s_acc = wmma_f16(qa1, kb1, s_acc);
      S[nt] = s_acc;
    }
    // ---- scale by sqrt(D)=32 and mask ----
#pragma unroll
    for (int nt = 0; nt < 4; ++nt) {
      int sc = s0 + nt * 16 + col;
#pragma unroll
      for (int e = 0; e < 8; ++e) {
        int qr = q0 + qw + e + rh;
        float val = S[nt][e] * 32.0f;
        int mk = mask[(size_t)qr * WR + sc];
        S[nt][e] = (mk == 0) ? -3.0e38f : val;
      }
    }
    // ---- online softmax ----
    float alpha[8];
#pragma unroll
    for (int e = 0; e < 8; ++e) {
      float r = fmaxf(fmaxf(S[0][e], S[1][e]), fmaxf(S[2][e], S[3][e]));
      r = redmax16(r);
      float mn = fmaxf(m_i[e], r);
      alpha[e] = __expf(m_i[e] - mn);
      m_i[e] = mn;
    }
#pragma unroll
    for (int j = 0; j < 4; ++j)
#pragma unroll
      for (int e = 0; e < 8; ++e) o[j][e] *= alpha[e];

    float rsum[8];
#pragma unroll
    for (int e = 0; e < 8; ++e) rsum[e] = 0.0f;
#pragma unroll
    for (int nt = 0; nt < 4; ++nt) {
#pragma unroll
      for (int e = 0; e < 8; ++e) {
        float p = __expf(S[nt][e] - m_i[e]);
        rsum[e] += p;
        Ps[(qw + e + rh) * PS + nt * 16 + col] = (_Float16)p;  // C/D -> A reshape
      }
    }
#pragma unroll
    for (int e = 0; e < 8; ++e) l_i[e] = l_i[e] * alpha[e] + redsum16(rsum[e]);

    // ---- O += P V  (P reloaded from per-wave LDS region as A frags) ----
    v16h pa0 = frag_a(&Ps[(qw + col) * PS + 0], lh);
    v16h pa1 = frag_a(&Ps[(qw + col) * PS + 32], lh);
#pragma unroll
    for (int nt = 0; nt < 4; ++nt) {
      v16h vb0 = frag_b(&Vts[(nt * 16 + col) * VS + 0], lh);
      v16h vb1 = frag_b(&Vts[(nt * 16 + col) * VS + 32], lh);
      o[nt] = wmma_f16(pa0, vb0, o[nt]);
      o[nt] = wmma_f16(pa1, vb1, o[nt]);
    }
  }

  // ---- normalize and store (f16, [b][r][h*DH+dh]) ----
#pragma unroll
  for (int e = 0; e < 8; ++e) {
    float inv = 1.0f / l_i[e];
    int qr = q0 + qw + e + rh;
#pragma unroll
    for (int nt = 0; nt < 4; ++nt) {
      int dh = nt * 16 + col;
      oh[base + (size_t)qr * WD + dh] = (_Float16)(o[nt][e] * inv);
    }
  }
}

// ---------------------------------------------------------------------------
extern "C" void kernel_launch(void* const* d_in, const int* in_sizes, int n_in,
                              void* d_out, int out_size, void* d_ws, size_t ws_size,
                              hipStream_t stream) {
  const float* x   = (const float*)d_in[0];
  const int*  mask = (const int*)d_in[1];
  const float* Wv  = (const float*)d_in[2];
  const float* bv  = (const float*)d_in[3];
  const float* Wk  = (const float*)d_in[4];
  const float* bk  = (const float*)d_in[5];
  const float* Wq  = (const float*)d_in[6];
  const float* bq  = (const float*)d_in[7];
  const float* Wo  = (const float*)d_in[8];
  const float* bo  = (const float*)d_in[9];
  float* out = (float*)d_out;

  const int M = WB * WR;                      // 8192
  const size_t planeB = (size_t)M * WD * sizeof(_Float16);
  char* w = (char*)d_ws;
  _Float16* qh = (_Float16*)(w);
  _Float16* kh = (_Float16*)(w + planeB);
  _Float16* vh = (_Float16*)(w + 2 * planeB);
  _Float16* ah = (_Float16*)(w + 3 * planeB);

  dim3 blk(256);
  dim3 pgrid(WD / 128, M / 128);
  hipLaunchKernelGGL((gemm_wmma<float, true>), pgrid, blk, 0, stream,
                     x, Wq, bq, (void*)qh, M, WD, WD);
  hipLaunchKernelGGL((gemm_wmma<float, true>), pgrid, blk, 0, stream,
                     x, Wk, bk, (void*)kh, M, WD, WD);
  hipLaunchKernelGGL((gemm_wmma<float, true>), pgrid, blk, 0, stream,
                     x, Wv, bv, (void*)vh, M, WD, WD);

  dim3 agrid(WR / 128, WB * WH);
  hipLaunchKernelGGL(attn_kernel, agrid, blk, 0, stream, qh, kh, vh, mask, ah);

  dim3 ogrid(WDOUT / 128, M / 128);
  hipLaunchKernelGGL((gemm_wmma<_Float16, false>), ogrid, blk, 0, stream,
                     ah, Wo, bo, (void*)out, M, WDOUT, WD);
}
